// TransLayer_59897613910226
// MI455X (gfx1250) — matrix operand order repaired and checked by reading, not compiled, not run on hardware
//
#include <hip/hip_runtime.h>
#include <hip/hip_bf16.h>

// ---------------------------------------------------------------------------
// Nystrom attention layer for MI455X (gfx1250).
// All matmuls: v_wmma_f32_16x16x32_bf16 (wave32), NT operands loaded as b128
// vectors in the ISA 7.12.2 register layouts; f32 softmax / pinv statistics.
// Every GEMM but k_gemm_pinv is in NT form (B stored transposed) so all operand fragments are
// contiguous 16-byte runs per lane.
// ---------------------------------------------------------------------------

typedef __bf16 bf16_t;
typedef __attribute__((ext_vector_type(16))) __bf16        v16bf;
typedef __attribute__((ext_vector_type(8)))  float         v8f;
typedef __attribute__((ext_vector_type(4)))  unsigned int  v4u;
typedef __attribute__((ext_vector_type(8)))  unsigned int  v8u;
typedef __attribute__((ext_vector_type(2)))  float         v2f;

#define SDIM   512
#define DH      64
#define NSEQ  8192
#define BATCH    8
#define ML     256
#define RESK    33

static __device__ __forceinline__ float  bf2f(bf16_t v) { return (float)v; }
static __device__ __forceinline__ bf16_t f2bf(float v)  { return (bf16_t)v; }

static __device__ __forceinline__ v16bf pack2(v4u lo, v4u hi) {
  v8u u;
  u[0] = lo[0]; u[1] = lo[1]; u[2] = lo[2]; u[3] = lo[3];
  u[4] = hi[0]; u[5] = hi[1]; u[6] = hi[2]; u[7] = hi[3];
  return __builtin_bit_cast(v16bf, u);
}

// A tile 16x32 bf16 at (row0,k0), row-major stride lda (lda % 8 == 0).
// Lane L: row L%16, K-half L/16; both fragment halves are 16B contiguous.
static __device__ __forceinline__ v16bf load_a_bf16(const bf16_t* __restrict__ base,
                                                    int lda, int lane) {
  const int row = lane & 15, kh = lane >> 4;
  const bf16_t* p = base + (size_t)row * lda + kh * 8;
  v4u lo = *(const v4u*)(p);        // K = kh*8 + 0..7   (vgpr 0..3)
  v4u hi = *(const v4u*)(p + 16);   // K = 16+kh*8+0..7  (vgpr 4..7)
  return pack2(lo, hi);
}

// B tile 32x16 with logical B[k][n] = Bt[n][k], Bt row-major N x K stride ldb.
// Lane L: col L%16, K = kh*8 + 0..7 and 16+kh*8+0..7 (same order as A).
static __device__ __forceinline__ v16bf load_b_nt_bf16(const bf16_t* __restrict__ base,
                                                       int ldb, int lane) {
  const int col = lane & 15, kh = lane >> 4;
  const bf16_t* p = base + (size_t)col * ldb + kh * 8;
  v4u lo = *(const v4u*)(p);
  v4u hi = *(const v4u*)(p + 16);
  return pack2(lo, hi);
}

static __device__ __forceinline__ v8f wmma_bf16(v16bf a, v16bf b, v8f c) {
  return __builtin_amdgcn_wmma_f32_16x16x32_bf16(false, a, false, b, (short)0, c,
                                                 false, false);
}

// ---------------------------------------------------------------------------
// Weight conversion with transpose: w_qkv[512x192] -> wqkvT[192x512],
// w_out[64x512] -> woutT[512x64], both bf16.
// ---------------------------------------------------------------------------
__global__ void k_convert_weights(const float* __restrict__ wqkv,
                                  const float* __restrict__ wout,
                                  bf16_t* __restrict__ wqkvT,
                                  bf16_t* __restrict__ woutT) {
  const int n1 = SDIM * 3 * DH;
  int i = blockIdx.x * blockDim.x + threadIdx.x;
  if (i < n1) {
    int r = i / (3 * DH), c = i % (3 * DH);
    wqkvT[(size_t)c * SDIM + r] = f2bf(wqkv[i]);
  } else if (i < n1 + DH * SDIM) {
    int j = i - n1;
    int r = j / SDIM, c = j % SDIM;
    woutT[(size_t)c * DH + r] = f2bf(wout[j]);
  }
}

// Batched transpose + f32->bf16: dst[b][c][r] = src[b][r][c]
__global__ void k_transpose_f2b(const float* __restrict__ src,
                                bf16_t* __restrict__ dst, int rows, int cols) {
  const size_t gid = (size_t)blockIdx.x * blockDim.x + threadIdx.x;
  const int c = (int)(gid % cols);
  const size_t rb = gid / cols;
  const int r = (int)(rb % rows);
  const size_t b = rb / rows;
  dst[(b * cols + c) * rows + r] = f2bf(src[gid]);
}

// ---------------------------------------------------------------------------
// LayerNorm, one wave per row of 512, bf16 out
// ---------------------------------------------------------------------------
__global__ void k_layernorm(const float* __restrict__ x,
                            const float* __restrict__ w,
                            const float* __restrict__ bia,
                            bf16_t* __restrict__ xn) {
  const int lane = threadIdx.x & 31;
  const size_t row = (size_t)blockIdx.x * 8 + (threadIdx.x >> 5);
  const float* p = x + row * SDIM;
  float vals[16];
  float s = 0.f;
#pragma unroll
  for (int i = 0; i < 16; ++i) { vals[i] = p[i * 32 + lane]; s += vals[i]; }
#pragma unroll
  for (int m = 16; m >= 1; m >>= 1) s += __shfl_xor(s, m, 32);
  const float mu = s * (1.0f / SDIM);
  float vs = 0.f;
#pragma unroll
  for (int i = 0; i < 16; ++i) { float d = vals[i] - mu; vs += d * d; }
#pragma unroll
  for (int m = 16; m >= 1; m >>= 1) vs += __shfl_xor(vs, m, 32);
  const float inv = rsqrtf(vs * (1.0f / SDIM) + 1e-5f);
#pragma unroll
  for (int i = 0; i < 16; ++i) {
    const int c = i * 32 + lane;
    xn[row * SDIM + c] = f2bf((vals[i] - mu) * inv * w[c] + bia[c]);
  }
}

// ---------------------------------------------------------------------------
// QKV GEMM: one wave per 16-row block computes all 12 column tiles (A reused
// 12x from registers). Epilogue splits into q(bf16, scaled), k(bf16),
// v(f32 row-major) and v^T(bf16, [B][64][8192]) for the av GEMM.
// ---------------------------------------------------------------------------
__global__ void __launch_bounds__(256, 1)
k_qkv(const bf16_t* __restrict__ xn,
      const bf16_t* __restrict__ wqkvT,
      bf16_t* __restrict__ q, bf16_t* __restrict__ k,
      float* __restrict__ vf, bf16_t* __restrict__ vtb) {
  const int wid = blockIdx.x * (blockDim.x >> 5) + (threadIdx.x >> 5);
  const int lane = threadIdx.x & 31;
  if (wid >= BATCH * NSEQ / 16) return;   // 4096 row blocks
  const bf16_t* Ab = xn + (size_t)(wid * 16) * SDIM;
  v8f acc[12] = {};
  for (int kk = 0; kk < SDIM; kk += 32) {
    if (kk + 64 < SDIM) __builtin_prefetch(Ab + kk + 64, 0, 1);
    v16bf a = load_a_bf16(Ab + kk, SDIM, lane);
#pragma unroll
    for (int j = 0; j < 12; ++j) {
      v16bf bb = load_b_nt_bf16(wqkvT + (size_t)(j * 16) * SDIM + kk, SDIM, lane);
      acc[j] = wmma_bf16(a, bb, acc[j]);
    }
  }
  const int colb = lane & 15;
  const int rb0 = wid * 16 + ((lane >> 4) << 3);
#pragma unroll
  for (int j = 0; j < 12; ++j) {
    const int col = j * 16 + colb;
#pragma unroll
    for (int r = 0; r < 8; ++r) {
      const size_t row = (size_t)(rb0 + r);
      const float val = acc[j][r];
      if (col < DH) {
        q[row * DH + col] = f2bf(val * 0.125f);          // dim_head^-0.5
      } else if (col < 2 * DH) {
        k[row * DH + (col - DH)] = f2bf(val);
      } else {
        const int d = col - 2 * DH;
        vf[row * DH + d] = val;
        const int n = (int)(row & (NSEQ - 1));
        const size_t b = row >> 13;
        vtb[((size_t)b * DH + d) * NSEQ + n] = f2bf(val);
      }
    }
  }
}

// ---------------------------------------------------------------------------
// Landmark means over contiguous groups of 32 rows
// ---------------------------------------------------------------------------
__global__ void k_landmarks(const bf16_t* __restrict__ q,
                            const bf16_t* __restrict__ k,
                            bf16_t* __restrict__ ql, bf16_t* __restrict__ kl) {
  const int m = blockIdx.x;       // [0, BATCH*ML)
  const int d = threadIdx.x;      // [0, 64)
  const size_t base = (size_t)m * 32 * DH + d;
  float sq = 0.f, sk = 0.f;
#pragma unroll 8
  for (int i = 0; i < 32; ++i) {
    sq += bf2f(q[base + (size_t)i * DH]);
    sk += bf2f(k[base + (size_t)i * DH]);
  }
  ql[(size_t)m * DH + d] = f2bf(sq * (1.0f / 32.0f));
  kl[(size_t)m * DH + d] = f2bf(sk * (1.0f / 32.0f));
}

// ---------------------------------------------------------------------------
// Generic batched NT GEMM, NB=4 column tiles per wave (A fragment reused 4x):
// C[MxNc] = A[MxK] @ Bt[NcxK]^T. Optional f32 / bf16 row-major outputs.
// ---------------------------------------------------------------------------
template <int NB>
__global__ void __launch_bounds__(256, 1)
k_gemm_ntb(const bf16_t* __restrict__ A, size_t sA, int lda,
           const bf16_t* __restrict__ Bt, size_t sB, int ldb,
           float* __restrict__ Cf, bf16_t* __restrict__ Cb,
           size_t sC, int ldc,
           int Nc, int K, int tilesPerBatch, int totalWaves) {
  const int wid = blockIdx.x * (blockDim.x >> 5) + (threadIdx.x >> 5);
  if (wid >= totalWaves) return;
  const int lane = threadIdx.x & 31;
  const int b = wid / tilesPerBatch;
  const int t = wid % tilesPerBatch;
  const int ngrp = Nc / (16 * NB);
  const int mt = t / ngrp, ng = t % ngrp;
  const bf16_t* Ab = A + (size_t)b * sA + (size_t)(mt * 16) * lda;
  const bf16_t* Bb = Bt + (size_t)b * sB + (size_t)(ng * 16 * NB) * ldb;
  v8f acc[NB] = {};
  for (int kk = 0; kk < K; kk += 32) {
    if (kk + 64 < K) __builtin_prefetch(Ab + kk + 64, 0, 1);
    v16bf a = load_a_bf16(Ab + kk, lda, lane);
#pragma unroll
    for (int j = 0; j < NB; ++j) {
      v16bf bb = load_b_nt_bf16(Bb + (size_t)(j * 16) * ldb + kk, ldb, lane);
      acc[j] = wmma_bf16(a, bb, acc[j]);
    }
  }
  const int colb = lane & 15;
  const int rb0 = mt * 16 + ((lane >> 4) << 3);
  const size_t cbase = (size_t)b * sC;
#pragma unroll
  for (int j = 0; j < NB; ++j) {
    const int col = (ng * NB + j) * 16 + colb;
#pragma unroll
    for (int r = 0; r < 8; ++r) {
      const size_t off = cbase + (size_t)(rb0 + r) * ldc + col;
      if (Cf) Cf[off] = acc[j][r];
      if (Cb) Cb[off] = f2bf(acc[j][r]);
    }
  }
}

// ---------------------------------------------------------------------------
// Row softmax (f32 src); optional f32 and bf16 outputs
// ---------------------------------------------------------------------------
__global__ void k_softmax(const float* __restrict__ src,
                          float* __restrict__ dstf,
                          bf16_t* __restrict__ dstb, int rowlen) {
  __shared__ float red[256];
  const size_t row = blockIdx.x;
  const float* p = src + row * (size_t)rowlen;
  const int tid = threadIdx.x;
  float m = -3.4e38f;
  for (int i = tid; i < rowlen; i += 256) m = fmaxf(m, p[i]);
  red[tid] = m; __syncthreads();
  for (int s = 128; s > 0; s >>= 1) {
    if (tid < s) red[tid] = fmaxf(red[tid], red[tid + s]);
    __syncthreads();
  }
  m = red[0]; __syncthreads();
  float sum = 0.f;
  for (int i = tid; i < rowlen; i += 256) sum += __expf(p[i] - m);
  red[tid] = sum; __syncthreads();
  for (int s = 128; s > 0; s >>= 1) {
    if (tid < s) red[tid] += red[tid + s];
    __syncthreads();
  }
  const float inv = 1.0f / red[0];
  for (int i = tid; i < rowlen; i += 256) {
    const float e = __expf(p[i] - m) * inv;
    const size_t off = row * (size_t)rowlen + i;
    if (dstf) dstf[off] = e;
    if (dstb) dstb[off] = f2bf(e);
  }
}

// ---------------------------------------------------------------------------
// Pinv: per-batch scale = 1/(max row-abs-sum * max col-abs-sum)
// ---------------------------------------------------------------------------
__global__ void k_pinv_scale(const float* __restrict__ x, float* __restrict__ scal) {
  __shared__ float rmax[256], cmax[256];
  const int b = blockIdx.x, t = threadIdx.x;
  const float* p = x + (size_t)b * (ML * ML);
  float rs = 0.f, cs = 0.f;
  for (int j = 0; j < ML; ++j) {
    rs += fabsf(p[(size_t)t * ML + j]);
    cs += fabsf(p[(size_t)j * ML + t]);
  }
  rmax[t] = rs; cmax[t] = cs; __syncthreads();
  for (int s = 128; s > 0; s >>= 1) {
    if (t < s) { rmax[t] = fmaxf(rmax[t], rmax[t + s]); cmax[t] = fmaxf(cmax[t], cmax[t + s]); }
    __syncthreads();
  }
  if (t == 0) scal[b] = 1.0f / (rmax[0] * cmax[0]);
}

__global__ void k_transpose_scale(const float* __restrict__ x,
                                  const float* __restrict__ scal,
                                  float* __restrict__ z) {
  const size_t gid = (size_t)blockIdx.x * blockDim.x + threadIdx.x; // B*256*256
  const int b = (int)(gid >> 16);
  const int r = (int)((gid >> 8) & 255);
  const int c = (int)(gid & 255);
  z[gid] = x[(size_t)b * (ML * ML) + (size_t)c * ML + r] * scal[b];
}

// Pinv iteration GEMM: D = cI*I + eA*A + s*(A@B), 256^3, f32 in/out,
// inline bf16 conversion feeding v_wmma_f32_16x16x32_bf16.
__global__ void __launch_bounds__(256, 1)
k_gemm_pinv(const float* __restrict__ A, const float* __restrict__ Bm,
            float* __restrict__ D, float cI, float eA, float s) {
  const int wid = blockIdx.x * (blockDim.x >> 5) + (threadIdx.x >> 5);
  const int lane = threadIdx.x & 31;
  const int b = wid >> 8;           // 256 tiles per batch
  const int t = wid & 255;
  const int mt = t >> 4, nt = t & 15;
  const float* Ab = A + (size_t)b * (ML * ML) + (size_t)(mt * 16) * ML;
  const float* Bb = Bm + (size_t)b * (ML * ML) + nt * 16;
  v8f acc = {};
  const int row16 = lane & 15, kh = lane >> 4;
  for (int kk = 0; kk < ML; kk += 32) {
    v16bf a, bb;
    const float* pa = Ab + (size_t)row16 * ML + kk;
#pragma unroll
    for (int v = 0; v < 8; ++v) {
      const int k = ((v >= 4) ? 16 : 0) + kh * 8 + (v & 3) * 2;
      v2f pr = *(const v2f*)(pa + k);
      a[2 * v]     = f2bf(pr[0]);
      a[2 * v + 1] = f2bf(pr[1]);
    }
    const float* pb = Bb + (size_t)(kk + kh * 8) * ML + row16;
#pragma unroll
    for (int v = 0; v < 8; ++v) {
      bb[2 * v]     = f2bf(pb[(size_t)(((v >= 4) ? 16 : 0) + (v & 3) * 2) * ML]);
      bb[2 * v + 1] = f2bf(pb[(size_t)(((v >= 4) ? 16 : 0) + (v & 3) * 2 + 1) * ML]);
    }
    acc = wmma_bf16(a, bb, acc);
  }
  const int col = nt * 16 + (lane & 15);
  const int rbase = mt * 16 + ((lane >> 4) << 3);
  const float* Ae = A + (size_t)b * (ML * ML);
  float* Db = D + (size_t)b * (ML * ML);
#pragma unroll
  for (int r = 0; r < 8; ++r) {
    const int row = rbase + r;
    float val = s * acc[r] + eA * Ae[(size_t)row * ML + col];
    if (row == col) val += cI;
    Db[(size_t)row * ML + col] = val;
  }
}

// ---------------------------------------------------------------------------
// Residual depthwise conv over sequence (kernel 33) + add; f32 + bf16 out
// ---------------------------------------------------------------------------
__global__ void k_resadd(const float* __restrict__ outmat,
                         const float* __restrict__ vf,
                         const float* __restrict__ wres,
                         float* __restrict__ sumf, bf16_t* __restrict__ sumb) {
  __shared__ float wk[RESK];
  if (threadIdx.x < RESK) wk[threadIdx.x] = wres[threadIdx.x];
  __syncthreads();
  const size_t gid = (size_t)blockIdx.x * blockDim.x + threadIdx.x; // B*N*64
  const int d = (int)(gid & (DH - 1));
  const size_t nd = gid >> 6;
  const int n = (int)(nd & (NSEQ - 1));
  const size_t b = nd >> 13;
  float acc = outmat[gid];
  const float* vb = vf + (b << 13) * DH;
#pragma unroll
  for (int j = 0; j < RESK; ++j) {
    const int nn = n + j - (RESK / 2);
    if ((unsigned)nn < (unsigned)NSEQ) acc += wk[j] * vb[(size_t)nn * DH + d];
  }
  sumf[gid] = acc;
  sumb[gid] = f2bf(acc);
}

// ---------------------------------------------------------------------------
// Final: out = proj + x + b_out  (proj already in d_out from GEMM)
// ---------------------------------------------------------------------------
__global__ void k_final(float* __restrict__ out, const float* __restrict__ x,
                        const float* __restrict__ bout) {
  const size_t gid = (size_t)blockIdx.x * blockDim.x + threadIdx.x;
  out[gid] = out[gid] + x[gid] + bout[gid & (SDIM - 1)];
}

// ---------------------------------------------------------------------------
extern "C" void kernel_launch(void* const* d_in, const int* in_sizes, int n_in,
                              void* d_out, int out_size, void* d_ws, size_t ws_size,
                              hipStream_t stream) {
  (void)in_sizes; (void)n_in; (void)out_size; (void)ws_size;
  const float* x     = (const float*)d_in[0];
  const float* ln_w  = (const float*)d_in[1];
  const float* ln_b  = (const float*)d_in[2];
  const float* w_qkv = (const float*)d_in[3];
  const float* w_out = (const float*)d_in[4];
  const float* b_out = (const float*)d_in[5];
  const float* w_res = (const float*)d_in[6];
  float* out = (float*)d_out;

  char* ws = (char*)d_ws;
  size_t off = 0;
  auto alloc = [&](size_t bytes) { size_t o = off; off += (bytes + 255) & ~(size_t)255; return o; };

  const size_t ROWS = (size_t)BATCH * NSEQ;                 // 65536
  const size_t o_xn    = alloc(ROWS * SDIM * 2);            // 64MB; reused as simbuf f32
  const size_t o_wqkvT = alloc(SDIM * 3 * DH * 2);
  const size_t o_woutT = alloc(DH * SDIM * 2);
  const size_t o_q     = alloc(ROWS * DH * 2);
  const size_t o_k     = alloc(ROWS * DH * 2);
  const size_t o_vf    = alloc(ROWS * DH * 4);
  const size_t o_vtb   = alloc(ROWS * DH * 2);              // v^T bf16 [B][64][8192]
  const size_t o_ql    = alloc((size_t)BATCH * ML * DH * 2);
  const size_t o_kl    = alloc((size_t)BATCH * ML * DH * 2);
  const size_t o_attn1 = alloc(ROWS * ML * 2);              // 32MB; reused: outmat f32 + outsum bf16
  const size_t o_attn3 = alloc(ROWS * ML * 2);              // 32MB; reused: tmp1 bf16
  const size_t o_sim2  = alloc((size_t)BATCH * ML * ML * 4);
  const size_t o_Z     = alloc((size_t)BATCH * ML * ML * 4);
  const size_t o_XZ    = alloc((size_t)BATCH * ML * ML * 4);
  const size_t o_U     = alloc((size_t)BATCH * ML * ML * 4);
  const size_t o_W     = alloc((size_t)BATCH * ML * ML * 4);
  const size_t o_Z2    = alloc((size_t)BATCH * ML * ML * 4);
  const size_t o_zinvT = alloc((size_t)BATCH * ML * ML * 2);
  const size_t o_scal  = alloc(BATCH * 4);
  const size_t o_av    = alloc((size_t)BATCH * ML * DH * 4);
  const size_t o_avT   = alloc((size_t)BATCH * ML * DH * 2);

  bf16_t* xn    = (bf16_t*)(ws + o_xn);
  float*  simbf = (float*)(ws + o_xn);                      // reuse after QKV
  bf16_t* wqkvT = (bf16_t*)(ws + o_wqkvT);
  bf16_t* woutT = (bf16_t*)(ws + o_woutT);
  bf16_t* qb    = (bf16_t*)(ws + o_q);
  bf16_t* kb    = (bf16_t*)(ws + o_k);
  float*  vf    = (float*)(ws + o_vf);
  bf16_t* vtb   = (bf16_t*)(ws + o_vtb);
  bf16_t* ql    = (bf16_t*)(ws + o_ql);
  bf16_t* kl    = (bf16_t*)(ws + o_kl);
  bf16_t* attn1 = (bf16_t*)(ws + o_attn1);
  bf16_t* attn3 = (bf16_t*)(ws + o_attn3);
  bf16_t* tmp1b = (bf16_t*)(ws + o_attn3);                  // reuse after av
  float*  outmat= (float*)(ws + o_attn1);                   // reuse after tmp1
  bf16_t* outsb = (bf16_t*)(ws + o_attn1 + ROWS * DH * 4);
  float*  sim2  = (float*)(ws + o_sim2);
  float*  Zbuf  = (float*)(ws + o_Z);
  float*  XZ    = (float*)(ws + o_XZ);
  float*  Ubuf  = (float*)(ws + o_U);
  float*  Wbuf  = (float*)(ws + o_W);
  float*  Z2    = (float*)(ws + o_Z2);
  bf16_t* zinvT = (bf16_t*)(ws + o_zinvT);
  float*  scal  = (float*)(ws + o_scal);
  float*  av    = (float*)(ws + o_av);
  bf16_t* avT   = (bf16_t*)(ws + o_avT);

  const dim3 B256(256);

  // 0) weights -> bf16 (transposed)
  k_convert_weights<<<dim3(512), B256, 0, stream>>>(w_qkv, w_out, wqkvT, woutT);
  // 1) layernorm
  k_layernorm<<<dim3(ROWS / 8), B256, 0, stream>>>(x, ln_w, ln_b, xn);
  // 2) qkv gemm + split (+ v^T)
  k_qkv<<<dim3(4096 / 8), B256, 0, stream>>>(xn, wqkvT, qb, kb, vf, vtb);
  // 3) landmarks
  k_landmarks<<<dim3(BATCH * ML), dim3(64), 0, stream>>>(qb, kb, ql, kl);
  // 4) sim2 = ql @ kl^T  [B,256,256]
  k_gemm_ntb<4><<<dim3(64), B256, 0, stream>>>(ql, (size_t)ML * DH, DH,
                                               kl, (size_t)ML * DH, DH,
                                               sim2, nullptr, (size_t)ML * ML, ML,
                                               ML, DH, 64, 512);
  // 5) softmax(sim2) -> attn2 f32 in place
  k_softmax<<<dim3(BATCH * ML), B256, 0, stream>>>(sim2, sim2, nullptr, ML);
  // 6) pinv init: Z = attn2^T / (max row-sum * max col-sum)
  k_pinv_scale<<<dim3(BATCH), B256, 0, stream>>>(sim2, scal);
  k_transpose_scale<<<dim3((BATCH * ML * ML) / 256), B256, 0, stream>>>(sim2, scal, Zbuf);
  // 7) six Newton-Schulz style iterations (4 WMMA GEMMs each)
  float* zc = Zbuf; float* za = Z2;
  for (int it = 0; it < 6; ++it) {
    k_gemm_pinv<<<dim3(256), B256, 0, stream>>>(sim2, zc, XZ, 0.f, 0.f, 1.f);
    k_gemm_pinv<<<dim3(256), B256, 0, stream>>>(XZ, XZ, Ubuf, 0.f, 7.f, -1.f);
    k_gemm_pinv<<<dim3(256), B256, 0, stream>>>(XZ, Ubuf, Wbuf, 0.f, 15.f, -1.f);
    k_gemm_pinv<<<dim3(256), B256, 0, stream>>>(zc, Wbuf, za, 0.f, 3.25f, -0.25f);
    float* t = zc; zc = za; za = t;
  }
  // Z^-1 transposed bf16 for the NT GEMM
  k_transpose_f2b<<<dim3((BATCH * ML * ML) / 256), B256, 0, stream>>>(zc, zinvT, ML, ML);
  // 8) sim1 = q @ kl^T [B,8192,256] -> softmax -> attn1 bf16
  k_gemm_ntb<4><<<dim3(2048), B256, 0, stream>>>(qb, (size_t)NSEQ * DH, DH,
                                                 kl, (size_t)ML * DH, DH,
                                                 simbf, nullptr, (size_t)NSEQ * ML, ML,
                                                 ML, DH, 2048, 16384);
  k_softmax<<<dim3(ROWS), B256, 0, stream>>>(simbf, nullptr, attn1, ML);
  // 9) sim3 = ql @ k^T [B,256,8192] -> softmax -> attn3 bf16
  k_gemm_ntb<4><<<dim3(2048), B256, 0, stream>>>(ql, (size_t)ML * DH, DH,
                                                 kb, (size_t)NSEQ * DH, DH,
                                                 simbf, nullptr, (size_t)ML * NSEQ, NSEQ,
                                                 NSEQ, DH, 2048, 16384);
  k_softmax<<<dim3(BATCH * ML), B256, 0, stream>>>(simbf, nullptr, attn3, NSEQ);
  // 10) av = attn3 @ v  [B,256,64] (B operand = v^T)
  k_gemm_ntb<4><<<dim3(16), B256, 0, stream>>>(attn3, (size_t)ML * NSEQ, NSEQ,
                                               vtb, (size_t)DH * NSEQ, NSEQ,
                                               av, nullptr, (size_t)ML * DH, DH,
                                               DH, NSEQ, 16, 128);
  k_transpose_f2b<<<dim3((BATCH * ML * DH) / 256), B256, 0, stream>>>(av, avT, ML, DH);
  // 11) tmp1 = attn1 @ Zinv  [B,8192,256] bf16 (reuses attn3 region)
  k_gemm_ntb<4><<<dim3(2048), B256, 0, stream>>>(attn1, (size_t)NSEQ * ML, ML,
                                                 zinvT, (size_t)ML * ML, ML,
                                                 nullptr, tmp1b, (size_t)NSEQ * ML, ML,
                                                 ML, ML, 2048, 16384);
  // 12) outmat = tmp1 @ av  [B,8192,64] f32 (reuses attn1 region)
  k_gemm_ntb<4><<<dim3(512), B256, 0, stream>>>(tmp1b, (size_t)NSEQ * ML, ML,
                                                avT, (size_t)DH * ML, ML,
                                                outmat, nullptr, (size_t)NSEQ * DH, DH,
                                                DH, ML, 512, 4096);
  // 13) + depthwise conv residual on v
  k_resadd<<<dim3((ROWS * DH) / 256), B256, 0, stream>>>(outmat, vf, w_res, outmat, outsb);
  // 14) projection: d_out = outsum @ w_out  [65536x64 @ 64x512]
  k_gemm_ntb<4><<<dim3(4096), B256, 0, stream>>>(outsb, 0, DH,
                                                 woutT, 0, DH,
                                                 out, nullptr, 0, SDIM,
                                                 SDIM, DH, 32768, 32768);
  // 15) + x + b_out
  k_final<<<dim3((ROWS * SDIM) / 256), B256, 0, stream>>>(out, x, b_out);
}
